// GQA_10952166604989
// MI455X (gfx1250) — matrix operand order repaired and checked by reading, not compiled, not run on hardware
//
#include <hip/hip_runtime.h>

#define DIMC 1024
#define NH   16
#define NKV  4
#define HD   64
#define BB   8
#define PP   1024

typedef __bf16 bf16;
typedef __attribute__((ext_vector_type(16))) __bf16 bf16x16;
typedef __attribute__((ext_vector_type(8)))  __bf16 bf16x8;
typedef __attribute__((ext_vector_type(8)))  float  f32x8;
typedef __attribute__((ext_vector_type(4)))  unsigned int u32x4;
typedef __attribute__((ext_vector_type(8)))  int i32x8;
typedef __attribute__((ext_vector_type(4)))  int i32x4;

__device__ __forceinline__ int lane_id() { return (int)(threadIdx.x & 31u); }

// A-fragment (16xK tile, K-step 32), ISA 7.12.2 "16-bit A-Matrix 16x32":
// lanes 0-15: row M=l, K = 0..7 & 16..23 ; lanes 16-31: row M=l-16, K = 8..15 & 24..31
__device__ __forceinline__ bf16x16 load_fragA(const bf16* __restrict__ p, int ld) {
  const int l = lane_id();
  const bf16* q = p + (l & 15) * ld + ((l >> 4) << 3);
  bf16x8 lo = *(const bf16x8*)(q);
  bf16x8 hi = *(const bf16x8*)(q + 16);
  bf16x16 f;
#pragma unroll
  for (int i = 0; i < 8; ++i) { f[i] = lo[i]; f[i + 8] = hi[i]; }
  return f;
}

// B-fragment (Kx16 tile, K-step 32): lanes 0-15: col N=l, K = 0..7 & 16..23;
// lanes 16-31: col N=l-16, K = 8..15 & 24..31. Memory is row-major [N][K] (W-style).
__device__ __forceinline__ bf16x16 load_fragB(const bf16* __restrict__ p, int ld) {
  const int l = lane_id();
  const bf16* q = p + (l & 15) * ld + ((l >> 4) << 3);
  bf16x8 lo = *(const bf16x8*)(q);
  bf16x8 hi = *(const bf16x8*)(q + 16);
  bf16x16 f;
#pragma unroll
  for (int i = 0; i < 8; ++i) { f[i] = lo[i]; f[i + 8] = hi[i]; }
  return f;
}

__device__ __forceinline__ f32x8 wmma_bf16(bf16x16 a, bf16x16 b, f32x8 c) {
  return __builtin_amdgcn_wmma_f32_16x16x32_bf16(false, a, false, b, (short)0, c,
                                                 false, false);
}

// Tensor Data Mover: 2D bf16 tile (tile_h rows x tile_w elems, global row stride
// stride_elems) -> contiguous LDS at lds_off. D# per ISA 08_async_tensor §8.3/8.4.
// 6-arg builtin on this toolchain: (g0, g1, g2, g3, g_extra, cpol).
__device__ __forceinline__ void tdm_load_2d_bf16(unsigned lds_off, const void* gptr,
                                                 unsigned tile_w, unsigned tile_h,
                                                 unsigned stride_elems) {
  unsigned long long ga = (unsigned long long)(size_t)gptr;
  u32x4 g0;
  g0[0] = 1u;                                        // count=1, user descriptor
  g0[1] = lds_off;                                   // lds_addr (bytes)
  g0[2] = (unsigned)(ga & 0xffffffffu);              // global_addr[31:0]
  g0[3] = (unsigned)((ga >> 32) & 0x01ffffffu) | (2u << 30);  // addr[56:32], type=2
  i32x8 g1;
  g1[0] = 1 << 16;                                   // data_size=1 -> 2 bytes
  g1[1] = (int)((tile_w & 0xffffu) << 16);           // tensor_dim0[15:0] = tile_w
  g1[2] = (int)((tile_w >> 16) | ((tile_h & 0xffffu) << 16));  // dim0 hi | dim1 lo
  g1[3] = (int)((tile_h >> 16) | (tile_w << 16));    // dim1 hi | tile_dim0
  g1[4] = (int)(tile_h & 0xffffu);                   // tile_dim1 (tile_dim2 = 0)
  g1[5] = (int)stride_elems;                         // tensor_dim0_stride[31:0]
  g1[6] = 0;
  g1[7] = 0;
  i32x4 z4 = {0, 0, 0, 0};
  i32x8 z8 = {0, 0, 0, 0, 0, 0, 0, 0};
  __builtin_amdgcn_tensor_load_to_lds(g0, g1, z4, z4, z8, 0);
}

// ---------------------------------------------------------------------------
__global__ void __launch_bounds__(256)
prep_x_kernel(const float* __restrict__ x, const int* __restrict__ perm,
              bf16* __restrict__ xb, bf16* __restrict__ xsh) {
  size_t i = (size_t)blockIdx.x * blockDim.x + threadIdx.x;  // over B*P*DIM
  xb[i] = (bf16)x[i];
  int d  = (int)(i & 63);
  int hh = (int)((i >> 6) & 15);
  size_t rowbase = i & ~(size_t)1023;
  xsh[i] = (bf16)x[rowbase + (size_t)perm[hh] * HD + d];
}

__global__ void __launch_bounds__(256)
cvt_f32_bf16_kernel(const float* __restrict__ in, bf16* __restrict__ out, int n) {
  int i = blockIdx.x * blockDim.x + threadIdx.x;
  if (i < n) out[i] = (bf16)in[i];
}

// ---------------------------------------------------------------------------
// C(MxN) = A(MxK) * W(NxK)^T, bf16 in / f32 accumulate.
// Each wave owns a 32x64 macro-tile: 2 A-frags + 4 B-frags -> 8 WMMAs per k-step.
// MODE 0: bf16 head-major [B][heads][P][HD]; MODE 1: bf16 [B][heads][HD][P];
// MODE 2: f32 row-major [M][N].
template <int MODE>
__global__ void __launch_bounds__(256)
gemm_bf16_kernel(const bf16* __restrict__ A, const bf16* __restrict__ W,
                 const float* __restrict__ bias, void* __restrict__ out,
                 int M, int N, int K, float scale, int heads) {
  const int wave = (int)(threadIdx.x >> 5);
  const int nmac_n = N >> 6;
  const int mac = blockIdx.x * 8 + wave;
  if (mac >= (M >> 5) * nmac_n) return;              // wave-uniform exit
  const int tm = (mac / nmac_n) << 5;
  const int tn = (mac % nmac_n) << 6;

  const bf16* pa = A + (size_t)tm * K;
  const bf16* pw = W + (size_t)tn * K;
  f32x8 acc[2][4];
#pragma unroll
  for (int i = 0; i < 2; ++i)
#pragma unroll
    for (int j = 0; j < 4; ++j) acc[i][j] = (f32x8){};

  for (int k0 = 0; k0 < K; k0 += 32) {
    __builtin_prefetch(pa + k0 + 128, 0, 3);         // global_prefetch_b8, near
    __builtin_prefetch(pw + k0 + 128, 0, 3);
    bf16x16 a0 = load_fragA(pa + k0, K);
    bf16x16 a1 = load_fragA(pa + (size_t)16 * K + k0, K);
    bf16x16 bfr[4];
#pragma unroll
    for (int t = 0; t < 4; ++t)
      bfr[t] = load_fragB(pw + (size_t)(t * 16) * K + k0, K);
#pragma unroll
    for (int t = 0; t < 4; ++t) {
      acc[0][t] = wmma_bf16(a0, bfr[t], acc[0][t]);
      acc[1][t] = wmma_bf16(a1, bfr[t], acc[1][t]);
    }
  }

  const int l  = lane_id();
  const int nl = l & 15;
  const int mb = (l >> 4) << 3;                      // C layout: VGPR j -> M = mb+j
#pragma unroll
  for (int mt = 0; mt < 2; ++mt) {
#pragma unroll
    for (int nt = 0; nt < 4; ++nt) {
      const int n = tn + nt * 16 + nl;
      const float bval = bias[n];
#pragma unroll
      for (int j = 0; j < 8; ++j) {
        const int m = tm + mt * 16 + mb + j;
        const float v = (acc[mt][nt][j] + bval) * scale;
        if (MODE == 0) {
          int b_ = m >> 10, p_ = m & 1023;
          size_t idx = (((size_t)(b_ * heads + (n >> 6)) * PP + p_) << 6) + (n & 63);
          ((bf16*)out)[idx] = (bf16)v;
        } else if (MODE == 1) {
          int b_ = m >> 10, p_ = m & 1023;
          size_t idx = (((size_t)(b_ * heads + (n >> 6)) * HD + (n & 63)) << 10) + p_;
          ((bf16*)out)[idx] = (bf16)v;
        } else {
          ((float*)out)[(size_t)m * N + n] = v;
        }
      }
    }
  }
}

// ---------------------------------------------------------------------------
// Flash-style GQA attention. Grid: B*NH*(P/128) blocks, 8 waves/block, one
// 16-query stripe per wave. K/V^T tiles are staged into double-buffered LDS by
// the Tensor Data Mover (wave 0), with the DMA for block kb+32 overlapped with
// compute on block kb. Online softmax staged through LDS.
__global__ void __launch_bounds__(256)
attn_kernel(const bf16* __restrict__ q,    // [B][NH][P][HD], pre-scaled
            const bf16* __restrict__ k,    // [B][NKV][P][HD]
            const bf16* __restrict__ vt,   // [B][NKV][HD][P]
            const int* __restrict__ perm,
            bf16* __restrict__ ao)         // [B][P][NH*HD]
{
  __shared__ float s_sc[8][16][34];
  __shared__ __align__(16) bf16 s_p[8][16][32];
  __shared__ __align__(16) bf16 s_k[2][32][HD];       // double-buffered key tiles
  __shared__ __align__(16) bf16 s_v[2][HD][32];       // double-buffered V^T tiles
  __shared__ float s_alpha[8][16];
  __shared__ float s_lsum[8][16];

  const int qt = (int)(blockIdx.x & 7);
  const int h  = (int)((blockIdx.x >> 3) & 15);
  const int b  = (int)(blockIdx.x >> 7);
  const int w  = (int)(threadIdx.x >> 5);
  const int l  = lane_id();
  const int nl = l & 15;
  const int mb = (l >> 4) << 3;
  const int r    = l & 15;
  const int half = l >> 4;

  int kv = 0;
  for (int i = 0; i < NH; ++i) if (perm[i] == h) kv = i >> 2;

  const int qr0 = qt * 128 + w * 16;
  const bf16* qp = q  + ((size_t)(b * NH + h) * PP + qr0) * HD;
  const bf16* kp = k  + (size_t)(b * NKV + kv) * PP * HD;
  const bf16* vp = vt + (size_t)(b * NKV + kv) * HD * PP;

  const bf16x16 aq0 = load_fragA(qp, HD);
  const bf16x16 aq1 = load_fragA(qp + 32, HD);

  f32x8 acc_o[4];
#pragma unroll
  for (int t = 0; t < 4; ++t) acc_o[t] = (f32x8){};
  float m_run = -3.0e38f, l_run = 0.f;

  unsigned lds_k[2], lds_v[2];
#pragma unroll
  for (int d = 0; d < 2; ++d) {
    lds_k[d] = (unsigned)(size_t)(&s_k[d][0][0]);
    lds_v[d] = (unsigned)(size_t)(&s_v[d][0][0]);
  }

  // prologue: stage block kb=0 into buffer 0
  if (w == 0) {
    tdm_load_2d_bf16(lds_k[0], kp, HD, 32, HD);
    tdm_load_2d_bf16(lds_v[0], vp, 32, HD, PP);
    __builtin_amdgcn_s_wait_tensorcnt(0);
  }
  __syncthreads();

  for (int kb = 0; kb < PP; kb += 32) {
    const int cur = (kb >> 5) & 1;
    // ---- issue TDM for the next block into the other buffer (overlapped)
    if (w == 0 && kb + 32 < PP) {
      tdm_load_2d_bf16(lds_k[cur ^ 1], kp + (size_t)(kb + 32) * HD, HD, 32, HD);
      tdm_load_2d_bf16(lds_v[cur ^ 1], vp + (kb + 32), 32, HD, PP);
    }

    // ---- scores: two 16x16 key tiles, HD=64 as two WMMA k-steps
    f32x8 s0 = {}, s1 = {};
    s0 = wmma_bf16(aq0, load_fragB(&s_k[cur][0][0], HD), s0);
    s0 = wmma_bf16(aq1, load_fragB(&s_k[cur][0][0] + 32, HD), s0);
    s1 = wmma_bf16(aq0, load_fragB(&s_k[cur][16][0], HD), s1);
    s1 = wmma_bf16(aq1, load_fragB(&s_k[cur][16][0] + 32, HD), s1);
#pragma unroll
    for (int j = 0; j < 8; ++j) {
      s_sc[w][mb + j][nl]      = s0[j];
      s_sc[w][mb + j][16 + nl] = s1[j];
    }
    __syncthreads();

    // ---- online softmax: lane (r, half) owns 16 of the 32 keys of row r
    float vals[16];
    float mx = -3.0e38f;
#pragma unroll
    for (int c = 0; c < 16; ++c) {
      float v = s_sc[w][r][half * 16 + c];
      vals[c] = v;
      mx = fmaxf(mx, v);
    }
    mx = fmaxf(mx, __shfl_xor(mx, 16, 32));
    const float mnew  = fmaxf(m_run, mx);
    const float alpha = __expf(m_run - mnew);
    float ls = 0.f;
#pragma unroll
    for (int c = 0; c < 16; ++c) {
      float pe = __expf(vals[c] - mnew);
      ls += pe;
      s_p[w][r][half * 16 + c] = (bf16)pe;
    }
    ls += __shfl_xor(ls, 16, 32);
    l_run = l_run * alpha + ls;
    m_run = mnew;
    if (half == 0) s_alpha[w][r] = alpha;
    __syncthreads();

    // ---- rescale accumulators, then O += P * V (4 dim-tiles, V from LDS)
#pragma unroll
    for (int j = 0; j < 8; ++j) {
      float al = s_alpha[w][mb + j];
#pragma unroll
      for (int t = 0; t < 4; ++t) acc_o[t][j] *= al;
    }
    const bf16x16 ap = load_fragA(&s_p[w][0][0], 32);
#pragma unroll
    for (int t = 0; t < 4; ++t) {
      bf16x16 bv = load_fragB(&s_v[cur][t * 16][0], 32);
      acc_o[t] = wmma_bf16(ap, bv, acc_o[t]);
    }

    // ---- next block's DMA must be complete before anyone touches it
    if (w == 0) __builtin_amdgcn_s_wait_tensorcnt(0);
    __syncthreads();
  }

  if (half == 0) s_lsum[w][r] = l_run;
  __syncthreads();
#pragma unroll
  for (int j = 0; j < 8; ++j) {
    const float inv = 1.f / s_lsum[w][mb + j];
    const int p_ = qr0 + mb + j;
#pragma unroll
    for (int t = 0; t < 4; ++t) {
      size_t idx = (((size_t)(b * PP + p_) * NH + h) << 6) + (t * 16 + nl);
      ao[idx] = (bf16)(acc_o[t][j] * inv);
    }
  }
}

// ---------------------------------------------------------------------------
extern "C" void kernel_launch(void* const* d_in, const int* in_sizes, int n_in,
                              void* d_out, int out_size, void* d_ws, size_t ws_size,
                              hipStream_t stream) {
  const float* x    = (const float*)d_in[0];
  const int*   perm = (const int*)d_in[1];
  const float* Wq   = (const float*)d_in[2];
  const float* bq   = (const float*)d_in[3];
  const float* Wk   = (const float*)d_in[4];
  const float* bk   = (const float*)d_in[5];
  const float* Wv   = (const float*)d_in[6];
  const float* bv   = (const float*)d_in[7];
  const float* Wp   = (const float*)d_in[8];
  const float* bp   = (const float*)d_in[9];
  float* out = (float*)d_out;

  char* ws = (char*)d_ws;
  size_t off = 0;
  auto alloc = [&](size_t bytes) -> void* {
    void* p = ws + off;
    off = (off + bytes + 255) & ~(size_t)255;
    return p;
  };
  const size_t NX = (size_t)BB * PP * DIMC;
  bf16* xb   = (bf16*)alloc(NX * 2);
  bf16* xsh  = (bf16*)alloc(NX * 2);
  bf16* wqb  = (bf16*)alloc((size_t)DIMC * DIMC * 2);
  bf16* wkb  = (bf16*)alloc((size_t)NKV * HD * DIMC * 2);
  bf16* wvb  = (bf16*)alloc((size_t)NKV * HD * DIMC * 2);
  bf16* wpb  = (bf16*)alloc((size_t)DIMC * DIMC * 2);
  bf16* qb   = (bf16*)alloc(NX * 2);
  bf16* kbuf = (bf16*)alloc((size_t)BB * NKV * PP * HD * 2);
  bf16* vtb  = (bf16*)alloc((size_t)BB * NKV * HD * PP * 2);
  bf16* aob  = (bf16*)alloc(NX * 2);

  cvt_f32_bf16_kernel<<<(DIMC * DIMC) / 256, 256, 0, stream>>>(Wq, wqb, DIMC * DIMC);
  cvt_f32_bf16_kernel<<<(NKV * HD * DIMC) / 256, 256, 0, stream>>>(Wk, wkb, NKV * HD * DIMC);
  cvt_f32_bf16_kernel<<<(NKV * HD * DIMC) / 256, 256, 0, stream>>>(Wv, wvb, NKV * HD * DIMC);
  cvt_f32_bf16_kernel<<<(DIMC * DIMC) / 256, 256, 0, stream>>>(Wp, wpb, DIMC * DIMC);

  prep_x_kernel<<<(int)(NX / 256), 256, 0, stream>>>(x, perm, xb, xsh);

  const int M = BB * PP;  // 8192
  // Q = x @ Wq^T + bq, scaled, head-major
  gemm_bf16_kernel<0><<<(M / 32) * (DIMC / 64) / 8, 256, 0, stream>>>(
      xb, wqb, bq, qb, M, DIMC, DIMC, 0.125f, NH);
  // K = x_sh @ Wk^T + bk
  gemm_bf16_kernel<0><<<(M / 32) * ((NKV * HD) / 64) / 8, 256, 0, stream>>>(
      xsh, wkb, bk, kbuf, M, NKV * HD, DIMC, 1.0f, NKV);
  // V^T = (x_sh @ Wv^T + bv), transposed per kv head
  gemm_bf16_kernel<1><<<(M / 32) * ((NKV * HD) / 64) / 8, 256, 0, stream>>>(
      xsh, wvb, bv, vtb, M, NKV * HD, DIMC, 1.0f, NKV);

  attn_kernel<<<BB * NH * (PP / 128), 256, 0, stream>>>(qb, kbuf, vtb, perm, aob);

  // final projection to f32 output
  gemm_bf16_kernel<2><<<(M / 32) * (DIMC / 64) / 8, 256, 0, stream>>>(
      aob, wpb, bp, out, M, DIMC, DIMC, 1.0f, NH);
}
